// EnhancedTransformerLayer_20057497272873
// MI455X (gfx1250) — matrix operand order repaired and checked by reading, not compiled, not run on hardware
//
#include <hip/hip_runtime.h>
#include <math.h>

// ---------------------------------------------------------------------------
// Types for WMMA (gfx1250, wave32)
// ---------------------------------------------------------------------------
typedef __attribute__((ext_vector_type(16))) __bf16 v16bf;
typedef __attribute__((ext_vector_type(8)))  float  v8f;
typedef __attribute__((ext_vector_type(4)))  int    i32x4;

struct __align__(16) U4 { unsigned x, y, z, w; };
struct __align__(8)  U2 { unsigned x, y; };
union FragU { U4 q[2]; v16bf v; };
struct Zoff { int a[8]; };   // per-z A offsets (elements) for batched GEMM

#define LN_EPS 1e-5f

// Async global->LDS path (gfx1250) — confirmed present on this toolchain.
#if defined(__has_builtin)
#  if __has_builtin(__builtin_amdgcn_global_load_async_to_lds_b128)
#    define GEMM_ASYNC 1
#  endif
#endif
#ifndef GEMM_ASYNC
#  define GEMM_ASYNC 0
#endif

__device__ __forceinline__ void async_wait0() {
#if GEMM_ASYNC
#  if __has_builtin(__builtin_amdgcn_s_wait_asynccnt)
  __builtin_amdgcn_s_wait_asynccnt(0);
#  else
  asm volatile("s_wait_asynccnt 0" ::: "memory");
#  endif
#endif
}

__device__ __forceinline__ float wsum(float v) {
#pragma unroll
  for (int off = 16; off > 0; off >>= 1) v += __shfl_xor(v, off, 32);
  return v;
}

__device__ __forceinline__ float gelu_f(float x) {
  return 0.5f * x * (1.0f + erff(x * 0.70710678118654752f));
}

// cheap round-to-nearest (ties away) f32->bf16: 2 VALU/elem, packs 4 -> 8B
__device__ __forceinline__ U2 pack4(float4 f) {
  unsigned a = __float_as_uint(f.x) + 0x8000u;
  unsigned b = __float_as_uint(f.y) + 0x8000u;
  unsigned c = __float_as_uint(f.z) + 0x8000u;
  unsigned d = __float_as_uint(f.w) + 0x8000u;
  U2 r;
  r.x = (a >> 16) | (b & 0xffff0000u);
  r.y = (c >> 16) | (d & 0xffff0000u);
  return r;
}

// fp32 -> bf16 bulk convert (4 elems/thread), n4 = n/4
__global__ __launch_bounds__(256)
void cvt_bf16(const float* __restrict__ in, unsigned short* __restrict__ out, long n4) {
  long i = (long)blockIdx.x * 256 + threadIdx.x;
  if (i >= n4) return;
  float4 f = ((const float4*)in)[i];
  ((U2*)out)[i] = pack4(f);
}

// ---------------------------------------------------------------------------
// WMMA GEMM: C[M,N] = act( A[M,K](fp32) @ Wb[N,K](bf16)^T + bias[N] )
// Block 256 thr (8 waves), tile 128 x (32*NJ), BK=32, double-buffered LDS,
// software pipelined.  W staged via async global->LDS (raw bf16), A staged
// fp32->bf16 through VGPRs.  blockIdx.z batching: A += zo.a[z],
// Wb += z*wz, bias += z*bz, C += z*cz.
// Requires M%128==0, N%(32*NJ)==0, K%32==0.
// ---------------------------------------------------------------------------
template <int NJ>
__global__ __launch_bounds__(256)
void gemm_wmma(const float* __restrict__ A, int lda,
               const unsigned short* __restrict__ Wb, int ldw,
               const float* __restrict__ bias,
               float* __restrict__ C, int ldc,
               int K, int act,
               Zoff zo, long wz, long bz, long cz) {
  constexpr int BN   = 32 * NJ;    // 64 or 128
  constexpr int WCPT = BN / 64;    // 16B W-chunks per thread (1 or 2)
  __shared__ __align__(16) unsigned short sA[2][128 * 40]; // stride 40 bf16 (80B)
  __shared__ __align__(16) unsigned short sW[2][BN * 40];

  const int z = blockIdx.z;
  A    += zo.a[z];
  Wb   += (size_t)z * wz;
  bias += (size_t)z * bz;
  C    += (size_t)z * cz;

  const int t    = threadIdx.x;
  const int wave = t >> 5, lane = t & 31;
  const int wm = wave & 3, wn = wave >> 2;
  const int r16 = lane & 15, half = lane >> 4;
  const size_t m0 = (size_t)blockIdx.y * 128;
  const size_t n0 = (size_t)blockIdx.x * BN;

  v8f zero = {0.f, 0.f, 0.f, 0.f, 0.f, 0.f, 0.f, 0.f};
  v8f acc[2][NJ];
#pragma unroll
  for (int mi = 0; mi < 2; ++mi)
#pragma unroll
    for (int nj = 0; nj < NJ; ++nj) acc[mi][nj] = zero;

  float4 areg[4];
#if !GEMM_ASYNC
  U4 wreg[WCPT];
#endif

  auto stageA_load = [&](int k0) {
#pragma unroll
    for (int i = 0; i < 4; ++i) {
      int idx = t + (i << 8);
      int rr = idx >> 3, c4 = idx & 7;
      areg[i] = *(const float4*)(A + (m0 + rr) * (size_t)lda + k0 + c4 * 4);
    }
  };
  auto stageA_store = [&](int buf) {
#pragma unroll
    for (int i = 0; i < 4; ++i) {
      int idx = t + (i << 8);
      int rr = idx >> 3, c4 = idx & 7;
      *(U2*)(&sA[buf][rr * 40 + c4 * 4]) = pack4(areg[i]);
    }
  };
  // W tile: BN rows x 32 cols bf16 = BN*4 chunks of 16B (8 bf16).
  auto stageW_issue = [&](int k0, int buf) {
#pragma unroll
    for (int i = 0; i < WCPT; ++i) {
      int idx = t + (i << 8);
      int rr = idx >> 2, c16 = idx & 3;
      const unsigned short* g = Wb + (n0 + rr) * (size_t)ldw + k0 + c16 * 8;
#if GEMM_ASYNC
      __builtin_amdgcn_global_load_async_to_lds_b128(
          (__attribute__((address_space(1))) i32x4*)g,
          (__attribute__((address_space(3))) i32x4*)&sW[buf][rr * 40 + c16 * 8],
          0, 0);
#else
      wreg[i] = *(const U4*)g;
      (void)buf;
#endif
    }
  };
  auto stageW_commit = [&](int buf) {
#if !GEMM_ASYNC
#pragma unroll
    for (int i = 0; i < WCPT; ++i) {
      int idx = t + (i << 8);
      int rr = idx >> 2, c16 = idx & 3;
      *(U4*)(&sW[buf][rr * 40 + c16 * 8]) = wreg[i];
    }
#else
    (void)buf;
#endif
  };

  const int nk = K >> 5;
  // prologue: stage tile 0 into buffer 0
  stageA_load(0);
  stageW_issue(0, 0);
  stageA_store(0);
  stageW_commit(0);
  async_wait0();
  __syncthreads();

  for (int kt = 0; kt < nk; ++kt) {
    const int cur = kt & 1, nxt = cur ^ 1;
    const bool more = (kt + 1) < nk;
    if (more) {                       // issue next tile early (hide latency)
      stageW_issue((kt + 1) << 5, nxt);
      stageA_load((kt + 1) << 5);
    }

    // A: 16-bit A layout (ISA 7.12.2): VGPR0-3 K=(0..7)+8*half, VGPR4-7 +16.
    FragU afr[2], bfr[NJ];
#pragma unroll
    for (int mi = 0; mi < 2; ++mi) {
      int ar = wm * 32 + mi * 16 + r16;
      afr[mi].q[0] = *(const U4*)(&sA[cur][ar * 40 + 8 * half]);
      afr[mi].q[1] = *(const U4*)(&sA[cur][ar * 40 + 16 + 8 * half]);
    }
    // B: same K layout as A: VGPR0-3 K=(0..7)+8*half, VGPR4-7 +16.
#pragma unroll
    for (int nj = 0; nj < NJ; ++nj) {
      int br = wn * (16 * NJ) + nj * 16 + r16;
      bfr[nj].q[0] = *(const U4*)(&sW[cur][br * 40 + 8 * half]);
      bfr[nj].q[1] = *(const U4*)(&sW[cur][br * 40 + 16 + 8 * half]);
    }
#pragma unroll
    for (int mi = 0; mi < 2; ++mi)
#pragma unroll
      for (int nj = 0; nj < NJ; ++nj)
        acc[mi][nj] = __builtin_amdgcn_wmma_f32_16x16x32_bf16(
            false, afr[mi].v, false, bfr[nj].v, (short)0, acc[mi][nj],
            false, false);

    if (more) {
      stageA_store(nxt);
      stageW_commit(nxt);
      async_wait0();
    }
    __syncthreads();
  }

  // Epilogue. D layout: lane -> col = lane%16, VGPR r -> row = r + 8*(lane/16).
#pragma unroll
  for (int mi = 0; mi < 2; ++mi) {
#pragma unroll
    for (int nj = 0; nj < NJ; ++nj) {
      size_t crow = m0 + wm * 32 + mi * 16 + 8 * half;
      size_t ccol = n0 + wn * (16 * NJ) + nj * 16 + r16;
      float bv = bias[ccol];
#pragma unroll
      for (int r = 0; r < 8; ++r) {
        float v = acc[mi][nj][r] + bv;
        if (act == 1) v = gelu_f(v);
        C[(crow + r) * (size_t)ldc + ccol] = v;
      }
    }
  }
}

// ---------------------------------------------------------------------------
// Row LayerNorm (wave per row). out = act(ln(a [+ b])). ncols in {64,256}.
// ---------------------------------------------------------------------------
__global__ __launch_bounds__(256)
void ln_row(float* __restrict__ out, int ostride,
            const float* __restrict__ a, int astride,
            const float* __restrict__ b, int bstride,
            const float* __restrict__ gamma, const float* __restrict__ beta,
            int ncols, long nrows, int act) {
  long row = (long)blockIdx.x * 8 + (threadIdx.x >> 5);
  if (row >= nrows) return;
  int lane = threadIdx.x & 31;
  int nc = ncols >> 5;
  float v[8];
  for (int i = 0; i < nc; ++i) {
    int c = lane + (i << 5);
    float x = a[row * (size_t)astride + c];
    if (b) x += b[row * (size_t)bstride + c];
    v[i] = x;
  }
  float s = 0.f;
  for (int i = 0; i < nc; ++i) s += v[i];
  float mu = wsum(s) / (float)ncols;
  float var = 0.f;
  for (int i = 0; i < nc; ++i) { float d = v[i] - mu; var += d * d; }
  var = wsum(var) / (float)ncols;
  float inv = rsqrtf(var + LN_EPS);
  for (int i = 0; i < nc; ++i) {
    int c = lane + (i << 5);
    float y = (v[i] - mu) * inv * gamma[c] + beta[c];
    if (act == 1) y = gelu_f(y);
    out[row * (size_t)ostride + c] = y;
  }
}

// qln over all tokens: out = ln(a) with per-modality params (s = row % 5).
__global__ __launch_bounds__(256)
void ln_row_s(float* __restrict__ out, const float* __restrict__ a,
              const float* __restrict__ gamma5, const float* __restrict__ beta5,
              long nrows) {
  long row = (long)blockIdx.x * 8 + (threadIdx.x >> 5);
  if (row >= nrows) return;
  int lane = threadIdx.x & 31;
  int s = (int)(row % 5);
  const float* g  = gamma5 + s * 256;
  const float* bt = beta5  + s * 256;
  float v[8];
  for (int i = 0; i < 8; ++i) v[i] = a[row * 256 + lane + (i << 5)];
  float ss = 0.f;
  for (int i = 0; i < 8; ++i) ss += v[i];
  float mu = wsum(ss) / 256.f;
  float var = 0.f;
  for (int i = 0; i < 8; ++i) { float d = v[i] - mu; var += d * d; }
  var = wsum(var) / 256.f;
  float inv = rsqrtf(var + LN_EPS);
  for (int i = 0; i < 8; ++i) {
    int c = lane + (i << 5);
    out[row * 256 + c] = (v[i] - mu) * inv * g[c] + bt[c];
  }
}

// gates for all tokens: out[row,0:5] = softmax(a[row] @ W5[s]^T + b5[s]), s=row%5
__global__ __launch_bounds__(256)
void rowdot_s(float* __restrict__ out, const float* __restrict__ a,
              const float* __restrict__ W5, const float* __restrict__ b5,
              long nrows) {
  long row = (long)blockIdx.x * 8 + (threadIdx.x >> 5);
  if (row >= nrows) return;
  int lane = threadIdx.x & 31;
  int s = (int)(row % 5);
  const float* W = W5 + (size_t)s * 5 * 256;
  const float* bs = b5 + s * 5;
  float vals[5];
#pragma unroll
  for (int n = 0; n < 5; ++n) {
    float p = 0.f;
    for (int k = lane; k < 256; k += 32)
      p += a[row * 256 + k] * W[n * 256 + k];
    vals[n] = wsum(p) + bs[n];
  }
  float mx = vals[0];
#pragma unroll
  for (int n = 1; n < 5; ++n) mx = fmaxf(mx, vals[n]);
  float se = 0.f;
#pragma unroll
  for (int n = 0; n < 5; ++n) { vals[n] = __expf(vals[n] - mx); se += vals[n]; }
  float r = 1.f / se;
  if (lane == 0)
#pragma unroll
    for (int n = 0; n < 5; ++n) out[row * 5 + n] = vals[n] * r;
}

// generic small-N row dot (importance logits, N=1, K=64)
__global__ __launch_bounds__(256)
void rowdot(float* __restrict__ out,
            const float* __restrict__ a, int astride,
            const float* __restrict__ W, const float* __restrict__ bias,
            int K, int N, long nrows) {
  long row = (long)blockIdx.x * 8 + (threadIdx.x >> 5);
  if (row >= nrows) return;
  int lane = threadIdx.x & 31;
  float vals[4];
  for (int n = 0; n < N; ++n) {
    float p = 0.f;
    for (int k = lane; k < K; k += 32)
      p += a[row * (size_t)astride + k] * W[n * (size_t)K + k];
    vals[n] = wsum(p) + bias[n];
  }
  if (lane == 0)
    for (int n = 0; n < N; ++n) out[row * (size_t)N + n] = vals[n];
}

// ---------------------------------------------------------------------------
// MHA over S=5, D=32, H=8. One wave per (b,h); lane = d.
// ---------------------------------------------------------------------------
__global__ __launch_bounds__(256)
void attn5(const float* __restrict__ qkv, float* __restrict__ o, long nbh) {
  long w = ((long)blockIdx.x * 256 + threadIdx.x) >> 5;
  if (w >= nbh) return;
  int lane = threadIdx.x & 31;
  long bb = w >> 3;
  int h = (int)(w & 7);
  float q[5], k[5], vv[5];
#pragma unroll
  for (int s = 0; s < 5; ++s) {
    size_t base = ((size_t)bb * 5 + s) * 768 + h * 32 + lane;
    q[s] = qkv[base]; k[s] = qkv[base + 256]; vv[s] = qkv[base + 512];
  }
  float a[5][5];
  const float scale = 0.17677669529663687f; // 1/sqrt(32)
#pragma unroll
  for (int s = 0; s < 5; ++s)
#pragma unroll
    for (int tt = 0; tt < 5; ++tt)
      a[s][tt] = wsum(q[s] * k[tt]) * scale;
#pragma unroll
  for (int s = 0; s < 5; ++s) {
    float mx = a[s][0];
#pragma unroll
    for (int tt = 1; tt < 5; ++tt) mx = fmaxf(mx, a[s][tt]);
    float se = 0.f;
#pragma unroll
    for (int tt = 0; tt < 5; ++tt) { a[s][tt] = __expf(a[s][tt] - mx); se += a[s][tt]; }
    float rinv = 1.f / se;
    float os = 0.f;
#pragma unroll
    for (int tt = 0; tt < 5; ++tt) os += a[s][tt] * rinv * vv[tt];
    o[((size_t)bb * 5 + s) * 256 + h * 32 + lane] = os;
  }
}

// importance = softmax over S of logits, per batch. Thread per b.
__global__ __launch_bounds__(256)
void softmax_s(const float* __restrict__ logit, float* __restrict__ imp, long Bn) {
  long b = (long)blockIdx.x * 256 + threadIdx.x;
  if (b >= Bn) return;
  float v[5], mx = -1e30f;
  for (int s = 0; s < 5; ++s) { v[s] = logit[b * 5 + s]; mx = fmaxf(mx, v[s]); }
  float se = 0.f;
  for (int s = 0; s < 5; ++s) { v[s] = __expf(v[s] - mx); se += v[s]; }
  float r = 1.f / se;
  for (int s = 0; s < 5; ++s) imp[b * 5 + s] = v[s] * r;
}

// global_attn = ln(rw*attn*mw[s]*imp[b,s] + (1-rw)*src). Wave per row.
__global__ __launch_bounds__(256)
void ga_ln(const float* __restrict__ src, const float* __restrict__ attn,
           const float* __restrict__ imp, const float* __restrict__ mod_w,
           const float* __restrict__ res_w,
           const float* __restrict__ g, const float* __restrict__ bta,
           float* __restrict__ out, long nrows) {
  long row = (long)blockIdx.x * 8 + (threadIdx.x >> 5);
  if (row >= nrows) return;
  int lane = threadIdx.x & 31;
  int s = (int)(row % 5);
  long bb = row / 5;
  float mw[5], mx = -1e30f;
  for (int i = 0; i < 5; ++i) { mw[i] = mod_w[i]; mx = fmaxf(mx, mw[i]); }
  float se = 0.f;
  for (int i = 0; i < 5; ++i) { mw[i] = __expf(mw[i] - mx); se += mw[i]; }
  float rw = 1.f / (1.f + __expf(-res_w[0]));
  float scale = rw * (mw[s] / se) * imp[bb * 5 + s];
  float om = 1.f - rw;
  float v[8];
  for (int i = 0; i < 8; ++i) {
    int c = lane + (i << 5);
    v[i] = scale * attn[row * 256 + c] + om * src[row * 256 + c];
  }
  float ss = 0.f;
  for (int i = 0; i < 8; ++i) ss += v[i];
  float mu = wsum(ss) / 256.f;
  float var = 0.f;
  for (int i = 0; i < 8; ++i) { float d = v[i] - mu; var += d * d; }
  var = wsum(var) / 256.f;
  float inv = rsqrtf(var + LN_EPS);
  for (int i = 0; i < 8; ++i) {
    int c = lane + (i << 5);
    out[row * 256 + c] = (v[i] - mu) * inv * g[c] + bta[c];
  }
}

// ---------------------------------------------------------------------------
// Fused modal combine for one modality s (wave per batch row b):
//   acc = g0*self + sum_j g[1+j]*ln(q + raw_j, cln[j])
//   modal[b*5+s] = ln(q + acc, fln)
// ---------------------------------------------------------------------------
__global__ __launch_bounds__(256)
void modal_combine(const float* __restrict__ src,     // (M,256) token matrix
                   const float* __restrict__ selfO,   // (B,256) for this s
                   const float* __restrict__ crossR,  // 4 x (B,256), stride crs
                   const float* __restrict__ gates,   // (M,5)
                   const float* __restrict__ clng, const float* __restrict__ clnb, // 4x256
                   const float* __restrict__ flng, const float* __restrict__ flnb, // 256
                   float* __restrict__ modal,         // (M,256) token matrix
                   int s, long Bn, long crs) {
  long b = (long)blockIdx.x * 8 + (threadIdx.x >> 5);
  if (b >= Bn) return;
  int lane = threadIdx.x & 31;
  long row = b * 5 + s;
  float g[5];
#pragma unroll
  for (int n = 0; n < 5; ++n) g[n] = gates[row * 5 + n];
  float q[8], acc[8];
#pragma unroll
  for (int i = 0; i < 8; ++i) {
    int c = lane + (i << 5);
    q[i]   = src[row * 256 + c];
    acc[i] = g[0] * selfO[b * 256 + c];
  }
#pragma unroll
  for (int j = 0; j < 4; ++j) {
    float tv[8];
    float ssum = 0.f;
#pragma unroll
    for (int i = 0; i < 8; ++i) {
      int c = lane + (i << 5);
      tv[i] = q[i] + crossR[j * crs + b * 256 + c];
      ssum += tv[i];
    }
    float mu = wsum(ssum) / 256.f;
    float var = 0.f;
#pragma unroll
    for (int i = 0; i < 8; ++i) { float d = tv[i] - mu; var += d * d; }
    var = wsum(var) / 256.f;
    float inv = rsqrtf(var + LN_EPS);
#pragma unroll
    for (int i = 0; i < 8; ++i) {
      int c = lane + (i << 5);
      acc[i] += g[1 + j] * ((tv[i] - mu) * inv * clng[j * 256 + c] + clnb[j * 256 + c]);
    }
  }
  // final: ln(q + acc)
  float ssum = 0.f;
#pragma unroll
  for (int i = 0; i < 8; ++i) { acc[i] += q[i]; ssum += acc[i]; }
  float mu = wsum(ssum) / 256.f;
  float var = 0.f;
#pragma unroll
  for (int i = 0; i < 8; ++i) { float d = acc[i] - mu; var += d * d; }
  var = wsum(var) / 256.f;
  float inv = rsqrtf(var + LN_EPS);
#pragma unroll
  for (int i = 0; i < 8; ++i) {
    int c = lane + (i << 5);
    modal[row * 256 + c] = (acc[i] - mu) * inv * flng[c] + flnb[c];
  }
}

// fusion gate + combine + norm1. Wave per row.
__global__ __launch_bounds__(256)
void fuse_ln(const float* __restrict__ src, const float* __restrict__ ga,
             const float* __restrict__ modal,
             const float* __restrict__ fusW, const float* __restrict__ fusb,
             const float* __restrict__ n1g, const float* __restrict__ n1b,
             float* __restrict__ x1, long nrows) {
  long row = (long)blockIdx.x * 8 + (threadIdx.x >> 5);
  if (row >= nrows) return;
  int lane = threadIdx.x & 31;
  float gav[8], mdv[8];
  float p = 0.f;
  for (int i = 0; i < 8; ++i) {
    int c = lane + (i << 5);
    gav[i] = ga[row * 256 + c];
    mdv[i] = modal[row * 256 + c];
    p += gav[i] * fusW[c] + mdv[i] * fusW[256 + c];
  }
  float dot = wsum(p) + fusb[0];
  float gg = 1.f / (1.f + __expf(-dot));
  float v[8];
  for (int i = 0; i < 8; ++i) {
    int c = lane + (i << 5);
    v[i] = src[row * 256 + c] + gg * gav[i] + (1.f - gg) * mdv[i];
  }
  float ss = 0.f;
  for (int i = 0; i < 8; ++i) ss += v[i];
  float mu = wsum(ss) / 256.f;
  float var = 0.f;
  for (int i = 0; i < 8; ++i) { float d = v[i] - mu; var += d * d; }
  var = wsum(var) / 256.f;
  float inv = rsqrtf(var + LN_EPS);
  for (int i = 0; i < 8; ++i) {
    int c = lane + (i << 5);
    x1[row * 256 + c] = (v[i] - mu) * inv * n1g[c] + n1b[c];
  }
}

// ---------------------------------------------------------------------------
// Host-side orchestration
// ---------------------------------------------------------------------------
extern "C" void kernel_launch(void* const* d_in, const int* in_sizes, int n_in,
                              void* d_out, int out_size, void* d_ws, size_t ws_size,
                              hipStream_t stream) {
  (void)in_sizes; (void)n_in; (void)out_size; (void)ws_size;

  const float* src      = (const float*)d_in[0];
  const float* sa_Wqkv  = (const float*)d_in[1];
  const float* sa_bqkv  = (const float*)d_in[2];
  const float* sa_Wo    = (const float*)d_in[3];
  const float* sa_bo    = (const float*)d_in[4];
  const float* sa_mod_w = (const float*)d_in[5];
  const float* imp_W1   = (const float*)d_in[6];
  const float* imp_b1   = (const float*)d_in[7];
  const float* imp_ln1g = (const float*)d_in[8];
  const float* imp_ln1b = (const float*)d_in[9];
  const float* imp_W2   = (const float*)d_in[10];
  const float* imp_b2   = (const float*)d_in[11];
  const float* imp_ln2g = (const float*)d_in[12];
  const float* imp_ln2b = (const float*)d_in[13];
  const float* imp_W3   = (const float*)d_in[14];
  const float* imp_b3   = (const float*)d_in[15];
  const float* sa_res_w = (const float*)d_in[16];
  const float* sa_ng    = (const float*)d_in[17];
  const float* sa_nb    = (const float*)d_in[18];
  const float* gsWqkv   = (const float*)d_in[19];
  const float* gsbqkv   = (const float*)d_in[20];
  const float* gsWo     = (const float*)d_in[21];
  const float* gsbo     = (const float*)d_in[22];
  const float* gcWqkv   = (const float*)d_in[23];
  const float* gcbqkv   = (const float*)d_in[24];
  const float* gcWo     = (const float*)d_in[25];
  const float* gcbo     = (const float*)d_in[26];
  const float* gclng    = (const float*)d_in[27];
  const float* gclnb    = (const float*)d_in[28];
  const float* gglng    = (const float*)d_in[29];
  const float* gglnb    = (const float*)d_in[30];
  const float* ggW      = (const float*)d_in[31];
  const float* ggb      = (const float*)d_in[32];
  const float* gflng    = (const float*)d_in[33];
  const float* gflnb    = (const float*)d_in[34];
  const float* ffW1     = (const float*)d_in[35];
  const float* ffb1     = (const float*)d_in[36];
  const float* ffW2     = (const float*)d_in[37];
  const float* ffb2     = (const float*)d_in[38];
  const float* n1g      = (const float*)d_in[39];
  const float* n1b      = (const float*)d_in[40];
  const float* n2g      = (const float*)d_in[41];
  const float* n2b      = (const float*)d_in[42];
  const float* fusW     = (const float*)d_in[43];
  const float* fusb     = (const float*)d_in[44];

  const long M = 40960, Bb = 8192; // M = B*S

  float* ws     = (float*)d_ws;
  float* qkv    = ws;                   // M*1024 (qkv first, FF1 later)
  float* t256a  = qkv    + M * 1024;    // h1 / attno / qln / x1
  float* t256b  = t256a  + M * 256;     // attn / ff2
  float* gab    = t256b  + M * 256;     // global_attn
  float* modal  = gab    + M * 256;     // modal combined (M,256 token order)
  float* h2b    = modal  + M * 256;     // M*64
  float* logit  = h2b    + M * 64;      // M
  float* impb   = logit  + M;           // Bb*5 (pad 8)
  float* gatesM = impb   + Bb * 8;      // M*5 (pad 8)
  float* selfT  = gatesM + M * 8;       // (S,B,256) = M*256
  float* selfO  = selfT  + M * 256;     // (S,B,256) = M*256
  float* crossT = selfO  + M * 256;     // 4 x (B,256)
  float* crossR = crossT + 4 * Bb * 256;// 4 x (B,256)

  // bf16 weight copies (converted once per call)
  unsigned short* wb    = (unsigned short*)(crossR + 4 * Bb * 256);
  unsigned short* wbW1  = wb;                      // 256*256
  unsigned short* wbW2  = wbW1  + 65536;           // 64*256
  unsigned short* wbQKV = wbW2  + 16384;           // 768*256
  unsigned short* wbWo  = wbQKV + 196608;          // 256*256
  unsigned short* wbGS  = wbWo  + 65536;           // 5*768*256
  unsigned short* wbGSo = wbGS  + 983040;          // 5*256*256
  unsigned short* wbGC  = wbGSo + 327680;          // 20*768*256
  unsigned short* wbGCo = wbGC  + 3932160;         // 20*256*256
  unsigned short* wbF1  = wbGCo + 1310720;         // 1024*256
  unsigned short* wbF2  = wbF1  + 262144;          // 256*1024

  dim3 blk(256);
  const unsigned gM = (unsigned)(M  / 128);  // 320
  const unsigned gB = (unsigned)(Bb / 128);  // 64
  auto lnG = [](long rows) { return dim3((unsigned)((rows + 7) / 8)); };
  auto cvtG = [](long n) { return dim3((unsigned)((n / 4 + 255) / 256)); };

  Zoff Z0 = {};                               // no batching
  Zoff Zb = {};                               // A offset = z * Bb*256
  for (int zz = 0; zz < 8; ++zz) Zb.a[zz] = zz * (int)(Bb * 256);
  Zoff Zs = {};                               // A offset = z * 256 (src modality)
  for (int zz = 0; zz < 8; ++zz) Zs.a[zz] = zz * 256;

  // 0) convert weights to bf16 once
  cvt_bf16<<<cvtG(65536),   blk, 0, stream>>>(imp_W1,  wbW1,  65536 / 4);
  cvt_bf16<<<cvtG(16384),   blk, 0, stream>>>(imp_W2,  wbW2,  16384 / 4);
  cvt_bf16<<<cvtG(196608),  blk, 0, stream>>>(sa_Wqkv, wbQKV, 196608 / 4);
  cvt_bf16<<<cvtG(65536),   blk, 0, stream>>>(sa_Wo,   wbWo,  65536 / 4);
  cvt_bf16<<<cvtG(983040),  blk, 0, stream>>>(gsWqkv,  wbGS,  983040 / 4);
  cvt_bf16<<<cvtG(327680),  blk, 0, stream>>>(gsWo,    wbGSo, 327680 / 4);
  cvt_bf16<<<cvtG(3932160), blk, 0, stream>>>(gcWqkv,  wbGC,  3932160 / 4);
  cvt_bf16<<<cvtG(1310720), blk, 0, stream>>>(gcWo,    wbGCo, 1310720 / 4);
  cvt_bf16<<<cvtG(262144),  blk, 0, stream>>>(ffW1,    wbF1,  262144 / 4);
  cvt_bf16<<<cvtG(262144),  blk, 0, stream>>>(ffW2,    wbF2,  262144 / 4);

  // 1) h1 = gelu(ln(src @ imp_W1^T + b1))
  gemm_wmma<4><<<dim3(2, gM), blk, 0, stream>>>(src, 256, wbW1, 256, imp_b1, t256a, 256, 256, 0, Z0, 0, 0, 0);
  ln_row<<<lnG(M), blk, 0, stream>>>(t256a, 256, t256a, 256, nullptr, 0, imp_ln1g, imp_ln1b, 256, M, 1);
  // 2) h2 = gelu(ln(h1 @ imp_W2^T + b2))
  gemm_wmma<2><<<dim3(1, gM), blk, 0, stream>>>(t256a, 256, wbW2, 256, imp_b2, h2b, 64, 256, 0, Z0, 0, 0, 0);
  ln_row<<<lnG(M), blk, 0, stream>>>(h2b, 64, h2b, 64, nullptr, 0, imp_ln2g, imp_ln2b, 64, M, 1);
  // 3) importance = softmax_S(h2 @ imp_W3^T + b3)
  rowdot<<<lnG(M), blk, 0, stream>>>(logit, h2b, 64, imp_W3, imp_b3, 64, 1, M);
  softmax_s<<<dim3(32), blk, 0, stream>>>(logit, impb, Bb);
  // 4) qkv
  gemm_wmma<4><<<dim3(6, gM), blk, 0, stream>>>(src, 256, wbQKV, 256, sa_bqkv, qkv, 768, 256, 0, Z0, 0, 0, 0);
  // 5) attention (S=5)
  attn5<<<dim3(8192), blk, 0, stream>>>(qkv, t256a, Bb * 8);
  // 6) attn = o @ sa_Wo^T + bo
  gemm_wmma<4><<<dim3(2, gM), blk, 0, stream>>>(t256a, 256, wbWo, 256, sa_bo, t256b, 256, 256, 0, Z0, 0, 0, 0);
  // 7) global_attn
  ga_ln<<<lnG(M), blk, 0, stream>>>(src, t256b, impb, sa_mod_w, sa_res_w, sa_ng, sa_nb, gab, M);

  // 8) modal branch
  //    gates for all tokens: qln (t256a free now) then gated softmax
  ln_row_s<<<lnG(M), blk, 0, stream>>>(t256a, src, gglng, gglnb, M);
  rowdot_s<<<lnG(M), blk, 0, stream>>>(gatesM, t256a, ggW, ggb, M);
  //    self path, batched over all 5 modalities (z = s)
  gemm_wmma<4><<<dim3(2, gB, 5), blk, 0, stream>>>(
      src, 1280, wbGS + 512 * 256, 256, gsbqkv + 512, selfT, 256, 256, 0,
      Zs, 768 * 256, 768, Bb * 256);
  gemm_wmma<4><<<dim3(2, gB, 5), blk, 0, stream>>>(
      selfT, 256, wbGSo, 256, gsbo, selfO, 256, 256, 0,
      Zb, 65536, 256, Bb * 256);
  //    cross paths per modality, batched over j (z = j)
  static const int OT[5][4] = {{1,2,3,4},{0,2,3,4},{0,1,3,4},{0,1,2,4},{0,1,2,3}};
  for (int s = 0; s < 5; ++s) {
    Zoff Zc = {};
    for (int j = 0; j < 4; ++j) Zc.a[j] = OT[s][j] * 256;
    gemm_wmma<4><<<dim3(2, gB, 4), blk, 0, stream>>>(
        src, 1280, wbGC + ((size_t)s * 4 * 768 + 512) * 256, 256,
        gcbqkv + (size_t)s * 4 * 768 + 512, crossT, 256, 256, 0,
        Zc, 768 * 256, 768, Bb * 256);
    gemm_wmma<4><<<dim3(2, gB, 4), blk, 0, stream>>>(
        crossT, 256, wbGCo + (size_t)s * 4 * 65536, 256, gcbo + (size_t)s * 4 * 256,
        crossR, 256, 256, 0,
        Zb, 65536, 256, Bb * 256);
    modal_combine<<<lnG(Bb), blk, 0, stream>>>(
        src, selfO + (size_t)s * Bb * 256, crossR, gatesM,
        gclng + (size_t)s * 4 * 256, gclnb + (size_t)s * 4 * 256,
        gflng + s * 256, gflnb + s * 256, modal, s, Bb, Bb * 256);
  }

  // 9) fusion gate + combine + norm1 -> x1 (t256a)
  fuse_ln<<<lnG(M), blk, 0, stream>>>(src, gab, modal, fusW, fusb, n1g, n1b, t256a, M);
  // 10) FF (gelu fused into GEMM epilogue); FF1 reuses the qkv region
  gemm_wmma<4><<<dim3(8, gM), blk, 0, stream>>>(t256a, 256, wbF1, 256, ffb1, qkv, 1024, 256, 1, Z0, 0, 0, 0);
  gemm_wmma<4><<<dim3(2, gM), blk, 0, stream>>>(qkv, 1024, wbF2, 1024, ffb2, t256b, 256, 1024, 0, Z0, 0, 0, 0);
  // 11) out = ln(x1 + ff)
  ln_row<<<lnG(M), blk, 0, stream>>>((float*)d_out, 256, t256a, 256, t256b, 256,
                                     n2g, n2b, 256, M, 0);
}
